// BDHChess_57526791962754
// MI455X (gfx1250) — matrix operand order repaired and checked by reading, not compiled, not run on hardware
//
#include <hip/hip_runtime.h>
#include <hip/hip_bf16.h>
#include <math.h>

namespace {

constexpr int BB      = 512;   // batch
constexpr int DD      = 256;   // n_embd
constexpr int NHH     = 4;     // heads
constexpr int NNN     = 512;   // neurons per head
constexpr int NT      = 2048;  // total neurons
constexpr int KSTEPS  = 8;
constexpr int TB      = 16;    // batches per block  (== WMMA M)
constexpr int NTHREADS= 256;   // 8 wave32

typedef __attribute__((ext_vector_type(16))) __bf16 v16bf;
typedef __attribute__((ext_vector_type(8)))  __bf16 v8bf;
typedef __attribute__((ext_vector_type(8)))  float  v8f;

__device__ __forceinline__ v8f wmma_bf16(v16bf a, v16bf b, v8f c) {
  // D = A(16x32 bf16) * B(32x16 bf16) + C(16x16 f32)
  return __builtin_amdgcn_wmma_f32_16x16x32_bf16(false, a, false, b, (short)0, c, false, false);
}

__device__ __forceinline__ float gelu_exact(float x) {
  return 0.5f * x * (1.0f + erff(x * 0.7071067811865475f));
}

// ---- WMMA fragment loaders (ISA 7.12.2 layouts, wave32) ----
// A 16x32 (16-bit): lane m=lane&15 is row M; per-lane K values are two contiguous
// 8-element runs: {0..7}+8*half and {16..23}+8*half  -> two 16B vector LDS loads.
__device__ __forceinline__ v16bf ldA_bf(const __bf16* base, int ldr, int k0, int lane) {
  int half = lane >> 4, m = lane & 15;
  const __bf16* p = base + m * ldr + k0 + 8 * half;
  v8bf lo = *(const v8bf*)p;
  v8bf hi = *(const v8bf*)(p + 16);
  v16bf a;
  #pragma unroll
  for (int t = 0; t < 8; ++t) { a[t] = lo[t]; a[t + 8] = hi[t]; }
  return a;
}
__device__ __forceinline__ v16bf ldA_f32(const float* base, int ldr, int k0, int lane) {
  v16bf a; int half = lane >> 4, m = lane & 15;
  const float* p = base + m * ldr + k0 + 8 * half;
  #pragma unroll
  for (int t = 0; t < 8; ++t) { a[t] = (__bf16)p[t]; a[t + 8] = (__bf16)p[t + 16]; }
  return a;
}
// B fragment from pre-packed bf16 weights: one aligned 32B load per lane.
// Packed layout: tile column nt holds K/32 tiles; tile (nt,kt) = 32 lanes x 16 bf16.
__device__ __forceinline__ v16bf ldB_pk(const __bf16* P, int kt, int lane) {
  return *(const v16bf*)(P + (((size_t)kt * 32 + lane) << 4));
}
// D 16x16 f32: VGPR r -> row r + 8*(lane>=16), col = lane&15
__device__ __forceinline__ void stD_lds(float* buf, int ldr, int n0, v8f c, int lane) {
  int half = lane >> 4, n = n0 + (lane & 15);
  #pragma unroll
  for (int r = 0; r < 8; ++r) buf[(r + 8 * half) * ldr + n] = c[r];
}

// Row-wise LayerNorm over [TB][DD] f32 LDS buffer; 16 threads per row.
__device__ void layernorm_rows(float* buf, const float* g, const float* beta,
                               float* red, int tid) {
  int row = tid >> 4, sub = tid & 15;
  float s = 0.f;
  for (int c = sub; c < DD; c += 16) s += buf[row * DD + c];
  red[row * 16 + sub] = s;
  __syncthreads();
  float mu = 0.f;
  #pragma unroll
  for (int i = 0; i < 16; ++i) mu += red[row * 16 + i];
  mu *= (1.0f / DD);
  __syncthreads();
  float v = 0.f;
  for (int c = sub; c < DD; c += 16) { float d = buf[row * DD + c] - mu; v += d * d; }
  red[row * 16 + sub] = v;
  __syncthreads();
  float var = 0.f;
  #pragma unroll
  for (int i = 0; i < 16; ++i) var += red[row * 16 + i];
  var *= (1.0f / DD);
  float rstd = rsqrtf(var + 1e-5f);
  for (int c = sub; c < DD; c += 16)
    buf[row * DD + c] = (buf[row * DD + c] - mu) * rstd * g[c] + beta[c];
  __syncthreads();
}

// ===== weight repack: f32 KxN row-major -> bf16 WMMA B-fragment order =====
// chunk idx = (nt*(K/32)+kt)*32 + lane ; each chunk = 16 bf16 (32 bytes)
__global__ __launch_bounds__(256)
void repack_w(const float* __restrict__ src, __bf16* __restrict__ dst,
              int K, int N, int src_bstride, int dst_bstride) {
  int b = blockIdx.y;
  const float* W = src + (size_t)b * src_bstride;
  __bf16* P = dst + (size_t)b * dst_bstride;
  int ktiles = K >> 5;
  int nchunk = ktiles * (N >> 4) * 32;
  for (int idx = blockIdx.x * blockDim.x + threadIdx.x; idx < nchunk;
       idx += gridDim.x * blockDim.x) {
    int lane = idx & 31;
    int t2 = idx >> 5;
    int kt = t2 % ktiles;
    int nt = t2 / ktiles;
    int half = lane >> 4, n = nt * 16 + (lane & 15);
    int kbase = kt * 32 + 8 * half;
    v16bf v;
    #pragma unroll
    for (int t = 0; t < 16; ++t) v[t] = (__bf16)W[(size_t)(kbase + (t & 7) + ((t >> 3) << 4)) * N + n];
    *(v16bf*)(P + ((size_t)idx << 4)) = v;
  }
}

// dynamic LDS layout (bytes)
constexpr int OFF_XB  = 0;                       // x   bf16 [TB][NT]
constexpr int OFF_YB  = OFF_XB + TB * NT * 2;    // y   bf16 [TB][NT] (also boards staging)
constexpr int OFF_AB  = OFF_YB + TB * NT * 2;    // a / a_cond / z+bx / h / vh-hidden  f32 [TB][DD]
constexpr int OFF_ZB  = OFF_AB + TB * DD * 4;    // z / b                               f32 [TB][DD]
constexpr int OFF_BY  = OFF_ZB + TB * DD * 4;    // b_y f32 [TB][DD]
constexpr int OFF_BX  = OFF_BY + TB * DD * 4;    // b_x f32 [TB][DD]
constexpr int OFF_RED = OFF_BX + TB * DD * 4;    // reductions f32 [TB][16]
constexpr int OFF_CL  = OFF_RED + TB * 16 * 4;   // dot coeffs f32 [TB][NHH]
constexpr int OFF_SQ  = OFF_CL + TB * NHH * 4;   // sumsq f32 [TB][NHH]
constexpr int OFF_INV = OFF_SQ + TB * NHH * 4;   // inv norms
constexpr int SMEM_BYTES = OFF_INV + TB * NHH * 4;   // ~198.5 KB  (< 320 KB WGP LDS)

__global__ __launch_bounds__(NTHREADS)
void bdh_chess_fused(const float* __restrict__ boards,
                     const float* __restrict__ enc_b1,
                     const float* __restrict__ enc_ln_g, const float* __restrict__ enc_ln_b,
                     const float* __restrict__ enc_b2,
                     const float* __restrict__ xinit_b,
                     const float* __restrict__ lna_g, const float* __restrict__ lna_b,
                     const float* __restrict__ lnz_g, const float* __restrict__ lnz_b,
                     const float* __restrict__ log_damping,
                     const float* __restrict__ vh_b1,
                     const float* __restrict__ vh_w2, const float* __restrict__ vh_b2,
                     const __bf16* __restrict__ p_enc1, const __bf16* __restrict__ p_enc2,
                     const __bf16* __restrict__ p_xinit,
                     const __bf16* __restrict__ p_by, const __bf16* __restrict__ p_bx,
                     const __bf16* __restrict__ p_Dy, const __bf16* __restrict__ p_E,
                     const __bf16* __restrict__ p_Dx, const __bf16* __restrict__ p_vh1,
                     float* __restrict__ xg,    // [BB][NT]      current x (f32 master)
                     float* __restrict__ xwH,   // [K][BB][NT]   x_write history
                     float* __restrict__ zH,    // [K][BB][DD]   z history
                     float* __restrict__ out)   // [BB]
{
  extern __shared__ unsigned char smem[];
  __bf16* xb  = (__bf16*)(smem + OFF_XB);
  __bf16* yb  = (__bf16*)(smem + OFF_YB);
  float*  abuf= (float*)(smem + OFF_AB);
  float*  zbuf= (float*)(smem + OFF_ZB);
  float*  byb = (float*)(smem + OFF_BY);
  float*  bxb = (float*)(smem + OFF_BX);
  float*  red = (float*)(smem + OFF_RED);
  float*  cL  = (float*)(smem + OFF_CL);
  float*  sq  = (float*)(smem + OFF_SQ);
  float*  inv = (float*)(smem + OFF_INV);

  const int tid  = threadIdx.x;
  const int lane = tid & 31;
  const int wave = tid >> 5;
  const int b0   = blockIdx.x * TB;

  const float lam = 1.0f / (1.0f + expf(-log_damping[0]));

  // ================= Encoder =================
  for (int e = tid; e < TB * 768; e += NTHREADS) {
    int r = e / 768, c = e - r * 768;
    yb[r * 768 + c] = (__bf16)boards[(size_t)(b0 + r) * 768 + c];
  }
  __syncthreads();

  // h = gelu(flat @ enc_w1 + b1)  -> abuf    (M16 N256 K768)
  {
    int nta = wave * 2;
    const __bf16* Pa = p_enc1 + (size_t)nta * (768 * 16);
    const __bf16* Pb = Pa + (size_t)768 * 16;
    v8f c0 = {0,0,0,0,0,0,0,0}, c1 = {0,0,0,0,0,0,0,0};
    for (int kt = 0; kt < 24; ++kt) {
      v16bf a = ldA_bf(yb, 768, kt * 32, lane);
      c0 = wmma_bf16(a, ldB_pk(Pa, kt, lane), c0);
      c1 = wmma_bf16(a, ldB_pk(Pb, kt, lane), c1);
    }
    int half = lane >> 4;
    #pragma unroll
    for (int r = 0; r < 8; ++r) {
      int row = r + 8 * half;
      int na = nta * 16 + (lane & 15), nb = na + 16;
      abuf[row * DD + na] = gelu_exact(c0[r] + enc_b1[na]);
      abuf[row * DD + nb] = gelu_exact(c1[r] + enc_b1[nb]);
    }
  }
  __syncthreads();
  layernorm_rows(abuf, enc_ln_g, enc_ln_b, red, tid);

  // b = h @ enc_w2 + b2 -> zbuf   (M16 N256 K256)
  {
    v16bf aF[8];
    #pragma unroll
    for (int kk = 0; kk < 8; ++kk) aF[kk] = ldA_f32(abuf, DD, kk * 32, lane);
    for (int i = 0; i < 2; ++i) {
      int nt = wave * 2 + i;
      const __bf16* P = p_enc2 + (size_t)nt * (DD * 16);
      v8f c = {0,0,0,0,0,0,0,0};
      #pragma unroll
      for (int kk = 0; kk < 8; ++kk) c = wmma_bf16(aF[kk], ldB_pk(P, kk, lane), c);
      int half = lane >> 4, n = nt * 16 + (lane & 15);
      #pragma unroll
      for (int r = 0; r < 8; ++r) zbuf[(r + 8 * half) * DD + n] = c[r] + enc_b2[n];
    }
  }
  __syncthreads();

  // b_y, b_x, x0
  {
    v16bf aF[8];
    #pragma unroll
    for (int kk = 0; kk < 8; ++kk) aF[kk] = ldA_f32(zbuf, DD, kk * 32, lane);
    for (int i = 0; i < 2; ++i) {
      int nt = wave * 2 + i;
      const __bf16* Py = p_by + (size_t)nt * (DD * 16);
      const __bf16* Px = p_bx + (size_t)nt * (DD * 16);
      v8f cy = {0,0,0,0,0,0,0,0}, cx = {0,0,0,0,0,0,0,0};
      #pragma unroll
      for (int kk = 0; kk < 8; ++kk) {
        cy = wmma_bf16(aF[kk], ldB_pk(Py, kk, lane), cy);
        cx = wmma_bf16(aF[kk], ldB_pk(Px, kk, lane), cx);
      }
      stD_lds(byb, DD, nt * 16, cy, lane);
      stD_lds(bxb, DD, nt * 16, cx, lane);
    }
    for (int i = 0; i < 16; ++i) {
      int nt = wave * 16 + i;
      const __bf16* P = p_xinit + (size_t)nt * (DD * 16);
      __builtin_prefetch((const void*)(P + DD * 16), 0, 1);
      v8f c = {0,0,0,0,0,0,0,0};
      #pragma unroll
      for (int kk = 0; kk < 8; ++kk) c = wmma_bf16(aF[kk], ldB_pk(P, kk, lane), c);
      int half = lane >> 4, n = nt * 16 + (lane & 15);
      #pragma unroll
      for (int r = 0; r < 8; ++r) {
        int row = r + 8 * half;
        float v = fmaxf(c[r] + xinit_b[n], 0.f);
        xg[(size_t)(b0 + row) * NT + n] = v;
        xb[row * NT + n] = (__bf16)v;
      }
    }
  }
  __syncthreads();

  // ================= K thinking steps (rho kept in low-rank form) =================
  for (int k = 0; k < KSTEPS; ++k) {
    // a = (1-lam) * sum_{j<k} lam^{k-1-j} * (xw_j . x)_bh * z_j
    for (int e = tid; e < TB * DD; e += NTHREADS) abuf[e] = 0.f;
    __syncthreads();
    for (int j = 0; j < k; ++j) {
      if (tid < TB * NHH) cL[tid] = 0.f;
      __syncthreads();
      {
        int row = tid >> 4, sub = tid & 15;
        const float* xw = xwH + ((size_t)j * BB + b0 + row) * NT;
        const __bf16* xr = xb + row * NT;
        #pragma unroll
        for (int h = 0; h < NHH; ++h) {
          float p = 0.f;
          for (int n = sub; n < NNN; n += 16) {
            int nn = h * NNN + n;
            p += xw[nn] * (float)xr[nn];
          }
          atomicAdd(&cL[row * NHH + h], p);
        }
      }
      __syncthreads();
      float lp = 1.f;
      for (int t = 0; t < k - 1 - j; ++t) lp *= lam;
      float coef = (1.f - lam) * lp;
      for (int e = tid; e < TB * DD; e += NTHREADS) {
        int row = e >> 8, d = e & 255, h = d >> 6;
        abuf[e] += coef * cL[row * NHH + h] *
                   zH[((size_t)j * BB + b0 + row) * DD + d];
      }
      __syncthreads();
    }
    for (int e = tid; e < TB * DD; e += NTHREADS) abuf[e] += byb[e];
    __syncthreads();
    layernorm_rows(abuf, lna_g, lna_b, red, tid);

    // g = relu(a_cond @ D_y[h]); y = g * x -> yb (bf16)
    {
      v16bf aF[8];
      #pragma unroll
      for (int kk = 0; kk < 8; ++kk) aF[kk] = ldA_f32(abuf, DD, kk * 32, lane);
      for (int i = 0; i < 16; ++i) {
        int n0 = (wave * 16 + i) * 16;
        int head = n0 >> 9;
        const __bf16* P = p_Dy + (size_t)head * (DD * NNN)
                               + (size_t)((n0 & (NNN - 1)) >> 4) * (DD * 16);
        __builtin_prefetch((const void*)(P + DD * 16), 0, 1);
        v8f c = {0,0,0,0,0,0,0,0};
        #pragma unroll
        for (int kk = 0; kk < 8; ++kk) c = wmma_bf16(aF[kk], ldB_pk(P, kk, lane), c);
        int half = lane >> 4, n = n0 + (lane & 15);
        #pragma unroll
        for (int r = 0; r < 8; ++r) {
          int row = r + 8 * half;
          float gv = fmaxf(c[r], 0.f);
          yb[row * NT + n] = (__bf16)(gv * (float)xb[row * NT + n]);
        }
      }
    }
    __syncthreads();

    // z = LN(y @ E) -> zbuf   (M16 N256 K2048)
    {
      int nta = wave * 2;
      const __bf16* Pa = p_E + (size_t)nta * (NT * 16);
      const __bf16* Pb = Pa + (size_t)NT * 16;
      v8f c0 = {0,0,0,0,0,0,0,0}, c1 = {0,0,0,0,0,0,0,0};
      for (int kt = 0; kt < 64; ++kt) {
        v16bf a = ldA_bf(yb, NT, kt * 32, lane);
        c0 = wmma_bf16(a, ldB_pk(Pa, kt, lane), c0);
        c1 = wmma_bf16(a, ldB_pk(Pb, kt, lane), c1);
      }
      stD_lds(zbuf, DD, nta * 16, c0, lane);
      stD_lds(zbuf, DD, nta * 16 + 16, c1, lane);
    }
    __syncthreads();
    layernorm_rows(zbuf, lnz_g, lnz_b, red, tid);

    // dx = relu((z + b_x) @ D_x[h]); x += dx; norms
    for (int e = tid; e < TB * DD; e += NTHREADS) abuf[e] = zbuf[e] + bxb[e];
    if (tid < TB * NHH) sq[tid] = 0.f;
    __syncthreads();
    {
      v16bf aF[8];
      #pragma unroll
      for (int kk = 0; kk < 8; ++kk) aF[kk] = ldA_f32(abuf, DD, kk * 32, lane);
      for (int i = 0; i < 16; ++i) {
        int n0 = (wave * 16 + i) * 16;
        int head = n0 >> 9;
        const __bf16* P = p_Dx + (size_t)head * (DD * NNN)
                               + (size_t)((n0 & (NNN - 1)) >> 4) * (DD * 16);
        __builtin_prefetch((const void*)(P + DD * 16), 0, 1);
        v8f c = {0,0,0,0,0,0,0,0};
        #pragma unroll
        for (int kk = 0; kk < 8; ++kk) c = wmma_bf16(aF[kk], ldB_pk(P, kk, lane), c);
        int half = lane >> 4, n = n0 + (lane & 15), h = n >> 9;
        #pragma unroll
        for (int r = 0; r < 8; ++r) {
          int row = r + 8 * half;
          float dxv = fmaxf(c[r], 0.f);
          size_t gi = (size_t)(b0 + row) * NT + n;
          float xn = xg[gi] + dxv;
          xg[gi] = xn;
          xb[row * NT + n] = (__bf16)xn;
          atomicAdd(&sq[row * NHH + h], xn * xn);
        }
      }
    }
    __syncthreads();
    if (tid < TB * NHH) inv[tid] = 1.0f / (sqrtf(sq[tid]) + 1e-8f);
    __syncthreads();

    for (int e = tid; e < TB * NT; e += NTHREADS) {
      int row = e >> 11, n = e & (NT - 1), h = n >> 9;
      float xv = xg[(size_t)(b0 + row) * NT + n];
      xwH[((size_t)k * BB + b0 + row) * NT + n] = xv * inv[row * NHH + h];
    }
    for (int e = tid; e < TB * DD; e += NTHREADS) {
      int row = e >> 8, d = e & 255;
      zH[((size_t)k * BB + b0 + row) * DD + d] = zbuf[e];
    }
    __syncthreads();
  }

  // ================= Value head =================
  {
    v16bf aF[8];
    #pragma unroll
    for (int kk = 0; kk < 8; ++kk) aF[kk] = ldA_f32(zbuf, DD, kk * 32, lane);
    const __bf16* P = p_vh1 + (size_t)wave * (DD * 16);
    v8f c = {0,0,0,0,0,0,0,0};
    #pragma unroll
    for (int kk = 0; kk < 8; ++kk) c = wmma_bf16(aF[kk], ldB_pk(P, kk, lane), c);
    int half = lane >> 4, n = wave * 16 + (lane & 15);
    #pragma unroll
    for (int r = 0; r < 8; ++r)
      abuf[(r + 8 * half) * 128 + n] = gelu_exact(c[r] + vh_b1[n]);
  }
  __syncthreads();
  {
    int row = tid >> 4, sub = tid & 15;
    float p = 0.f;
    for (int i = sub; i < 128; i += 16) p += abuf[row * 128 + i] * vh_w2[i];
    red[row * 16 + sub] = p;
    __syncthreads();
    if (sub == 0) {
      float s = 0.f;
      #pragma unroll
      for (int i = 0; i < 16; ++i) s += red[row * 16 + i];
      out[b0 + row] = tanhf(s + vh_b2[0]);
    }
  }
}

} // namespace

extern "C" void kernel_launch(void* const* d_in, const int* in_sizes, int n_in,
                              void* d_out, int out_size, void* d_ws, size_t ws_size,
                              hipStream_t stream) {
  (void)in_sizes; (void)n_in; (void)out_size; (void)ws_size;
  const float* boards      = (const float*)d_in[0];
  const float* enc_w1      = (const float*)d_in[1];
  const float* enc_b1      = (const float*)d_in[2];
  const float* enc_ln_g    = (const float*)d_in[3];
  const float* enc_ln_b    = (const float*)d_in[4];
  const float* enc_w2      = (const float*)d_in[5];
  const float* enc_b2      = (const float*)d_in[6];
  const float* xinit_w     = (const float*)d_in[7];
  const float* xinit_b     = (const float*)d_in[8];
  const float* D_y         = (const float*)d_in[9];
  const float* E           = (const float*)d_in[10];
  const float* D_x         = (const float*)d_in[11];
  const float* By_w        = (const float*)d_in[12];
  const float* Bx_w        = (const float*)d_in[13];
  const float* lna_g       = (const float*)d_in[14];
  const float* lna_b       = (const float*)d_in[15];
  const float* lnz_g       = (const float*)d_in[16];
  const float* lnz_b       = (const float*)d_in[17];
  const float* log_damping = (const float*)d_in[18];
  const float* vh_w1       = (const float*)d_in[19];
  const float* vh_b1       = (const float*)d_in[20];
  const float* vh_w2       = (const float*)d_in[21];
  const float* vh_b2       = (const float*)d_in[22];

  // workspace (f32): histories + x master (~42 MB, L2-resident)
  float* xwH = (float*)d_ws;                                  // [K][BB][NT]
  float* zH  = xwH + (size_t)KSTEPS * BB * NT;                // [K][BB][DD]
  float* xg  = zH  + (size_t)KSTEPS * BB * DD;                // [BB][NT]
  // packed bf16 weights after the f32 region (~4.9 MB)
  __bf16* pbase = (__bf16*)(xg + (size_t)BB * NT);
  size_t o = 0;
  __bf16* p_enc1  = pbase + o; o += (size_t)768 * 256;
  __bf16* p_enc2  = pbase + o; o += (size_t)256 * 256;
  __bf16* p_xinit = pbase + o; o += (size_t)256 * 2048;
  __bf16* p_by    = pbase + o; o += (size_t)256 * 256;
  __bf16* p_bx    = pbase + o; o += (size_t)256 * 256;
  __bf16* p_Dy    = pbase + o; o += (size_t)NHH * 256 * 512;
  __bf16* p_E     = pbase + o; o += (size_t)2048 * 256;
  __bf16* p_Dx    = pbase + o; o += (size_t)NHH * 256 * 512;
  __bf16* p_vh1   = pbase + o; o += (size_t)256 * 128;

  auto rp = [&](const float* src, __bf16* dst, int K, int N, int nb,
                int sstride, int dstride) {
    int nchunk = (K / 32) * (N / 16) * 32;
    int blocks = (nchunk + 255) / 256;
    dim3 g(blocks, nb);
    repack_w<<<g, 256, 0, stream>>>(src, dst, K, N, sstride, dstride);
  };
  rp(enc_w1,  p_enc1,  768, 256,  1, 0, 0);
  rp(enc_w2,  p_enc2,  256, 256,  1, 0, 0);
  rp(xinit_w, p_xinit, 256, 2048, 1, 0, 0);
  rp(By_w,    p_by,    256, 256,  1, 0, 0);
  rp(Bx_w,    p_bx,    256, 256,  1, 0, 0);
  rp(D_y,     p_Dy,    256, 512,  NHH, 256 * 512, 256 * 512);
  rp(E,       p_E,     2048, 256, 1, 0, 0);
  rp(D_x,     p_Dx,    256, 512,  NHH, 256 * 512, 256 * 512);
  rp(vh_w1,   p_vh1,   256, 128,  1, 0, 0);

  dim3 grid(BB / TB), block(NTHREADS);
  bdh_chess_fused<<<grid, block, SMEM_BYTES, stream>>>(
      boards, enc_b1, enc_ln_g, enc_ln_b, enc_b2, xinit_b,
      lna_g, lna_b, lnz_g, lnz_b, log_damping,
      vh_b1, vh_w2, vh_b2,
      p_enc1, p_enc2, p_xinit, p_by, p_bx, p_Dy, p_E, p_Dx, p_vh1,
      xg, xwH, zH, (float*)d_out);
}
